// StackedDirectionalConv3d_14061722927566
// MI455X (gfx1250) — matrix operand order repaired and checked by reading, not compiled, not run on hardware
//
#include <hip/hip_runtime.h>

// ---- CDNA5 (gfx1250, wave32) WMMA types ----
typedef __attribute__((ext_vector_type(16))) __bf16          v16bf;
typedef __attribute__((ext_vector_type(8)))  float           v8f;
typedef __attribute__((ext_vector_type(8)))  unsigned short  u16x8;

union Frag { u16x8 h[2]; v16bf v; };

#define CIN    256
#define COUT   256
#define TRC    32768          // T*R*C = 32*32*32
#define M_TILE 256            // full COUT per block -> X tile read once per N-tile
#define N_TILE 128
#define NITER  56             // 7 direction groups * 8 K-chunks of 32

__device__ __forceinline__ float bfbits2f(unsigned short h) {
  unsigned u = ((unsigned)h) << 16;
  return __builtin_bit_cast(float, u);
}
struct BfPair { unsigned short h, l; };
// 2-term split: f ~= hi + lo, each bf16 (RNE via native convert)
__device__ __forceinline__ BfPair cvt_split(float f) {
  BfPair p;
  __bf16 hb = (__bf16)f;
  p.h = __builtin_bit_cast(unsigned short, hb);
  __bf16 lb = (__bf16)(f - bfbits2f(p.h));
  p.l = __builtin_bit_cast(unsigned short, lb);
  return p;
}

__global__ __launch_bounds__(512)
void sdconv3d_wmma(const float* __restrict__ x, const float* __restrict__ W,
                   float* __restrict__ out) {
  // Double-buffered bf16 hi/lo tiles, K-contiguous for b128 fragment loads.
  __shared__ unsigned short Ahi[2][M_TILE * 32];   // 32 KB
  __shared__ unsigned short Alo[2][M_TILE * 32];   // 32 KB
  __shared__ unsigned short Bhi[2][N_TILE * 32];   // 16 KB
  __shared__ unsigned short Blo[2][N_TILE * 32];   // 16 KB

  const int tid  = threadIdx.x;
  const int lane = tid & 31;
  const int wid  = tid >> 5;      // 16 waves
  const int wm   = wid & 3;       // 4 x 64-row slices of M
  const int wn   = wid >> 2;      // 4 x 32-col slices of N

  const int n0 = blockIdx.x * N_TILE;   // spatial tile base

  // B-loader geometry: 128 cols x 32 k, 8 strided k-loads per thread
  const int jn = tid & (N_TILE - 1);
  const int k8 = (tid >> 7) * 8;        // 0,8,16,24
  const int nn = n0 + jn;
  const int bb = nn >> 15;
  const int ss = nn & 32767;
  const int tq = ss >> 10, rq = (ss >> 5) & 31, cq = ss & 31;

  // A-loader geometry: 256 rows x 32 k, 16 floats per thread
  const int arow = tid >> 1;
  const int akh  = (tid & 1) * 16;

  v8f acc[4][2];
  {
    v8f z = {};
#pragma unroll
    for (int i = 0; i < 4; ++i) { acc[i][0] = z; acc[i][1] = z; }
  }

  float4 wst[4];   // staged W  (16 f32)
  float  xst[8];   // staged X  (8 f32)

  // ---- issue global loads for iteration `it` into stage registers ----
  auto issue_loads = [&](int it) {
    const int g  = it >> 3;
    const int kc = it & 7;
    const float* wp = W + (size_t)(g * COUT + arow) * CIN + kc * 32 + akh;
    wst[0] = *(const float4*)(wp + 0);
    wst[1] = *(const float4*)(wp + 4);
    wst[2] = *(const float4*)(wp + 8);
    wst[3] = *(const float4*)(wp + 12);

    int dtg = 0, drg = 0, dcg = 0;          // uniform (SALU) branch on g
    if      (g == 1) dtg = -1; else if (g == 2) dtg = 1;
    else if (g == 3) drg = -1; else if (g == 4) drg = 1;
    else if (g == 5) dcg = -1; else if (g == 6) dcg = 1;
    const int  tt = tq + dtg, rr = rq + drg, cc = cq + dcg;
    const bool valid = ((unsigned)tt < 32u) && ((unsigned)rr < 32u) && ((unsigned)cc < 32u);
    const int  src = (tt << 10) + (rr << 5) + cc;
    const float* xp = x + (size_t)(bb * CIN + kc * 32 + k8) * TRC + src;
#pragma unroll
    for (int kk = 0; kk < 8; ++kk) xst[kk] = valid ? xp[(size_t)kk * TRC] : 0.0f;
  };

  // ---- convert staged f32 -> bf16 hi/lo and store into LDS buffer ----
  auto store_tile = [&](int buf) {
    const float* tf = (const float*)wst;
    u16x8 h0, h1, l0, l1;
#pragma unroll
    for (int i = 0; i < 8; ++i) {
      BfPair p0 = cvt_split(tf[i]);
      BfPair p1 = cvt_split(tf[i + 8]);
      h0[i] = p0.h;  l0[i] = p0.l;
      h1[i] = p1.h;  l1[i] = p1.l;
    }
    const int ab = arow * 32 + akh;
    *(u16x8*)&Ahi[buf][ab]     = h0;  *(u16x8*)&Ahi[buf][ab + 8] = h1;
    *(u16x8*)&Alo[buf][ab]     = l0;  *(u16x8*)&Alo[buf][ab + 8] = l1;
    u16x8 xh, xl;
#pragma unroll
    for (int kk = 0; kk < 8; ++kk) {
      BfPair p = cvt_split(xst[kk]);
      xh[kk] = p.h;  xl[kk] = p.l;
    }
    *(u16x8*)&Bhi[buf][jn * 32 + k8] = xh;
    *(u16x8*)&Blo[buf][jn * 32 + k8] = xl;
  };

  // ---- 24 x v_wmma_f32_16x16x32_bf16 per wave per K-chunk ----
  auto compute = [&](int buf) {
    Frag bh[2], bl[2];
#pragma unroll
    for (int ns = 0; ns < 2; ++ns) {
      const int jj    = wn * 32 + ns * 16 + (lane & 15);
      const int bbase = jj * 32 + ((lane & 16) ? 8 : 0);
      bh[ns].h[0] = *(const u16x8*)&Bhi[buf][bbase];
      bh[ns].h[1] = *(const u16x8*)&Bhi[buf][bbase + 16];
      bl[ns].h[0] = *(const u16x8*)&Blo[buf][bbase];
      bl[ns].h[1] = *(const u16x8*)&Blo[buf][bbase + 16];
    }
#pragma unroll
    for (int ms = 0; ms < 4; ++ms) {
      const int row   = wm * 64 + ms * 16 + (lane & 15);
      const int abase = row * 32 + ((lane & 16) ? 8 : 0);
      Frag ah, al;
      ah.h[0] = *(const u16x8*)&Ahi[buf][abase];
      ah.h[1] = *(const u16x8*)&Ahi[buf][abase + 16];
      al.h[0] = *(const u16x8*)&Alo[buf][abase];
      al.h[1] = *(const u16x8*)&Alo[buf][abase + 16];
#pragma unroll
      for (int ns = 0; ns < 2; ++ns) {
        // 3xBF16 split: hh + hl + lh
        acc[ms][ns] = __builtin_amdgcn_wmma_f32_16x16x32_bf16(
            false, ah.v, false, bh[ns].v, (short)0, acc[ms][ns], false, false);
        acc[ms][ns] = __builtin_amdgcn_wmma_f32_16x16x32_bf16(
            false, ah.v, false, bl[ns].v, (short)0, acc[ms][ns], false, false);
        acc[ms][ns] = __builtin_amdgcn_wmma_f32_16x16x32_bf16(
            false, al.v, false, bh[ns].v, (short)0, acc[ms][ns], false, false);
      }
    }
  };

  // ---- software pipeline: global loads for it+1 overlap WMMA of it ----
  issue_loads(0);
  store_tile(0);
  __syncthreads();

  for (int it = 0; it < NITER; ++it) {
    const int cur = it & 1;
    if (it + 1 < NITER) issue_loads(it + 1);   // vmem in flight under WMMAs
    compute(cur);
    if (it + 1 < NITER) store_tile(cur ^ 1);   // waits loadcnt, fills next buf
    __syncthreads();                           // single barrier per iteration
  }

  // ---- store: D frag layout -> out[b, o, t, r, c] ----
  const int mhi = (lane & 16) ? 8 : 0;
#pragma unroll
  for (int ms = 0; ms < 4; ++ms) {
#pragma unroll
    for (int ns = 0; ns < 2; ++ns) {
      const int nf = n0 + wn * 32 + ns * 16 + (lane & 15);
      const int ob = nf >> 15;
      const int os = nf & 32767;
      float* op = out + (size_t)(ob * COUT) * TRC + os;
#pragma unroll
      for (int rv = 0; rv < 8; ++rv) {
        const int m = wm * 64 + ms * 16 + rv + mhi;
        op[(size_t)m * TRC] = acc[ms][ns][rv];
      }
    }
  }
}

extern "C" void kernel_launch(void* const* d_in, const int* in_sizes, int n_in,
                              void* d_out, int out_size, void* d_ws, size_t ws_size,
                              hipStream_t stream) {
  (void)in_sizes; (void)n_in; (void)out_size; (void)d_ws; (void)ws_size;
  const float* x = (const float*)d_in[0];   // (2, 256, 32, 32, 32) f32
  const float* W = (const float*)d_in[1];   // (1792, 256) f32
  float* out = (float*)d_out;               // (2, 256, 32, 32, 32) f32

  dim3 grid(65536 / N_TILE);                // 512 blocks
  dim3 block(512);                          // 16 wave32 waves
  hipLaunchKernelGGL(sdconv3d_wmma, grid, block, 0, stream, x, W, out);
}
